// HeteroGraphModel_45715631898884
// MI455X (gfx1250) — matrix operand order repaired and checked by reading, not compiled, not run on hardware
//
#include <hip/hip_runtime.h>
#include <hip/hip_bf16.h>

typedef _Float16 half_t;
typedef __attribute__((ext_vector_type(16))) _Float16 v16h;
typedef __attribute__((ext_vector_type(8)))  _Float16 v8h;
typedef __attribute__((ext_vector_type(4)))  _Float16 v4h;
typedef __attribute__((ext_vector_type(8)))  float    v8f;

#define D_FEAT 128
#define HIDDEN 128
#define OUTD   64
#define HEADS  4

#define SHUF16(lo, hi) __builtin_shufflevector(lo, hi, 0,1,2,3,4,5,6,7,8,9,10,11,12,13,14,15)

// ---------- order-preserving float<->uint encoding for atomicMax segment-max ----------
__device__ __forceinline__ unsigned enc_f32(float f) {
  int i = __float_as_int(f);
  return (i >= 0) ? ((unsigned)i | 0x80000000u) : ~(unsigned)i;
}
__device__ __forceinline__ float dec_f32(unsigned u) {
  int i = (u & 0x80000000u) ? (int)(u & 0x7FFFFFFFu) : (int)(~u);
  return __int_as_float(i);
}
static inline unsigned host_enc_f32(float f) {
  union { float f; int i; } u; u.f = f;
  return (u.i >= 0) ? ((unsigned)u.i | 0x80000000u) : ~(unsigned)u.i;
}

// ---------- tiny utility kernels ----------
__global__ void k_fill_f32(float* p, float v, long n) {
  long i = (long)blockIdx.x * blockDim.x + threadIdx.x;
  long s = (long)gridDim.x * blockDim.x;
  for (; i < n; i += s) p[i] = v;
}
__global__ void k_fill_u32(unsigned* p, unsigned v, long n) {
  long i = (long)blockIdx.x * blockDim.x + threadIdx.x;
  long s = (long)gridDim.x * blockDim.x;
  for (; i < n; i += s) p[i] = v;
}
__global__ void k_cast_f16(const float* __restrict__ in, half_t* __restrict__ out, long n) {
  long i = (long)blockIdx.x * blockDim.x + threadIdx.x;
  long s = (long)gridDim.x * blockDim.x;
  for (; i < n; i += s) out[i] = (half_t)in[i];
}

// ---------- pack f32 weights [K=128 x ncols] into WMMA B-fragment layout, fused f16 cast ----
// out[((ks*ntn + nt)*32 + lane)*16 + i] = B[(ks*32 + (lane/16)*8 + (i/8)*16 + i%8)*ldb + colbase + nt*16 + lane%16]
__global__ void k_pack_b(const float* __restrict__ B, int ldb, int colbase, int ncols,
                         half_t* __restrict__ out) {
  int ntn = ncols >> 4;
  int total = 4 * ntn * 512;
  int tid = blockIdx.x * blockDim.x + threadIdx.x;
  if (tid >= total) return;
  int i    = tid & 15;
  int lane = (tid >> 4) & 31;
  int tile = tid >> 9;
  int ks = tile / ntn, nt = tile % ntn;
  int k = ks * 32 + ((lane >> 4) & 1) * 8 + (i >> 3) * 16 + (i & 7);
  int c = colbase + nt * 16 + (lane & 15);
  out[tid] = (half_t)B[(size_t)k * ldb + c];
}

// Precontract attention vectors: wsa[k,h] = sum_c W_src[k, h*C+c] * att_src[h,c] (same for dst)
__global__ void k_att_combine(const float* __restrict__ Wsrc, const float* __restrict__ Wdst,
                              const float* __restrict__ att_s, const float* __restrict__ att_d,
                              float* __restrict__ wsa, float* __restrict__ wda) {
  int tid = threadIdx.x;
  if (tid >= HIDDEN * HEADS) return;
  int k = tid >> 2, h = tid & 3;
  float s = 0.f, d = 0.f;
  for (int c = 0; c < HIDDEN; ++c) {
    s += Wsrc[(size_t)k * (HEADS * HIDDEN) + h * HIDDEN + c] * att_s[h * HIDDEN + c];
    d += Wdst[(size_t)k * (HEADS * HIDDEN) + h * HIDDEN + c] * att_d[h * HIDDEN + c];
  }
  wsa[k * HEADS + h] = s;
  wda[k * HEADS + h] = d;
}

__device__ __forceinline__ v8f wmma_f16(v16h a, v16h b, v8f c) {
  return __builtin_amdgcn_wmma_f32_16x16x32_f16(false, a, false, b, (short)0, c, false, false);
}

// ---------- SAGE: edge scatter (wave per edge; 32 lanes x float4 = 128 ch) ----------
__global__ void k_sage_scatter(const float* __restrict__ x, const int* __restrict__ src,
                               const int* __restrict__ dst, float* __restrict__ agg,
                               float* __restrict__ cnt, int E) {
  int tid = blockIdx.x * blockDim.x + threadIdx.x;
  int e = tid >> 5, lane = tid & 31;
  if (e >= E) return;
  int s = src[e], d = dst[e];
  float4 v = ((const float4*)(x + (size_t)s * D_FEAT))[lane];
  float* ap = agg + (size_t)d * D_FEAT + lane * 4;
  atomicAdd(&ap[0], v.x);
  atomicAdd(&ap[1], v.y);
  atomicAdd(&ap[2], v.z);
  atomicAdd(&ap[3], v.w);
  if (lane == 0) atomicAdd(&cnt[d], 1.0f);
}

// ---------- SAGE GEMM: h = relu(mean @ W_l + x @ W_r + b); LDS-staged packed weights ----------
__global__ void __launch_bounds__(256) k_sage_gemm(
    const float* __restrict__ agg, const float* __restrict__ cnt,
    const half_t* __restrict__ x_h, const half_t* __restrict__ wl_p,
    const half_t* __restrict__ wr_p, const float* __restrict__ b_sage,
    half_t* __restrict__ h_out, int N) {
  __shared__ v8h sm[1024];               // [0..511] W_l tiles (this ks), [512..1023] W_r
  int tid = threadIdx.x;
  int waveid = tid >> 5, lane = tid & 31;
  int rt = blockIdx.x * 8 + waveid;
  int row0 = rt << 4;
  int g = lane >> 4;
  int rA = row0 + (lane & 15);
  if (rA >= N) rA = N - 1;
  float inv = 1.0f / fmaxf(cnt[rA], 1.0f);
  const float*  arow = agg + (size_t)rA * D_FEAT;
  const half_t* xrow = x_h + (size_t)rA * D_FEAT;

  v8f acc[8];
#pragma unroll
  for (int nt = 0; nt < 8; ++nt)
#pragma unroll
    for (int r = 0; r < 8; ++r) acc[nt][r] = 0.0f;

  for (int ks = 0; ks < 4; ++ks) {
    __syncthreads();
    const v8h* srcl = (const v8h*)wl_p + ks * 512;
    const v8h* srcr = (const v8h*)wr_p + ks * 512;
    sm[tid]       = srcl[tid];
    sm[tid + 256] = srcl[tid + 256];
    sm[tid + 512] = srcr[tid];
    sm[tid + 768] = srcr[tid + 256];
    __syncthreads();

    int kb = ks * 32 + g * 8;
    float fl[16];
    *(float4*)&fl[0]  = *(const float4*)(arow + kb);
    *(float4*)&fl[4]  = *(const float4*)(arow + kb + 4);
    *(float4*)&fl[8]  = *(const float4*)(arow + kb + 16);
    *(float4*)&fl[12] = *(const float4*)(arow + kb + 20);
    v16h a1;
#pragma unroll
    for (int i = 0; i < 16; ++i) a1[i] = (half_t)(fl[i] * inv);
    v8h xlo = *(const v8h*)(xrow + kb);
    v8h xhi = *(const v8h*)(xrow + kb + 16);
    v16h a2 = SHUF16(xlo, xhi);

#pragma unroll
    for (int nt = 0; nt < 8; ++nt) {
      int bi = nt * 64 + lane * 2;
      v16h bl = SHUF16(sm[bi], sm[bi + 1]);
      v16h br = SHUF16(sm[512 + bi], sm[512 + bi + 1]);
      acc[nt] = wmma_f16(a1, bl, acc[nt]);
      acc[nt] = wmma_f16(a2, br, acc[nt]);
    }
  }

  int colb = lane & 15;
  int rbase = row0 + g * 8;
#pragma unroll
  for (int nt = 0; nt < 8; ++nt) {
    int col = nt * 16 + colb;
    float bias = b_sage[col];
#pragma unroll
    for (int r = 0; r < 8; ++r) {
      int rr = rbase + r;
      if (rr < N) {
        float v = acc[nt][r] + bias;
        h_out[(size_t)rr * HIDDEN + col] = (half_t)fmaxf(v, 0.0f);
      }
    }
  }
}

// ---------- GAT GEMM: hs = h @ W_src -> f16 [N,4,128]; block = (rowgroup, 128-col strip) ----------
__global__ void __launch_bounds__(256) k_gat_gemm(
    const half_t* __restrict__ hA, const half_t* __restrict__ wsrc_p,
    half_t* __restrict__ hs, int N) {
  __shared__ v8h sm[512];                // 8 tiles of current (st, ks)
  int tid = threadIdx.x;
  int waveid = tid >> 5, lane = tid & 31;
  int st = blockIdx.x & 3;
  int rt = (blockIdx.x >> 2) * 8 + waveid;
  int row0 = rt << 4;
  int g = lane >> 4;
  int rA = row0 + (lane & 15);
  if (rA >= N) rA = N - 1;
  const half_t* arow = hA + (size_t)rA * HIDDEN;

  v8f acc[8];
#pragma unroll
  for (int nt = 0; nt < 8; ++nt)
#pragma unroll
    for (int r = 0; r < 8; ++r) acc[nt][r] = 0.0f;

  for (int ks = 0; ks < 4; ++ks) {
    __syncthreads();
    const v8h* src = (const v8h*)wsrc_p + (size_t)(st * 4 + ks) * 512;
    sm[tid]       = src[tid];
    sm[tid + 256] = src[tid + 256];
    __syncthreads();

    int kb = ks * 32 + g * 8;
    v8h alo = *(const v8h*)(arow + kb);
    v8h ahi = *(const v8h*)(arow + kb + 16);
    v16h a = SHUF16(alo, ahi);

#pragma unroll
    for (int nt = 0; nt < 8; ++nt) {
      int bi = nt * 64 + lane * 2;
      v16h b = SHUF16(sm[bi], sm[bi + 1]);
      acc[nt] = wmma_f16(a, b, acc[nt]);
    }
  }

  int colb = lane & 15;
  int rbase = row0 + g * 8;
#pragma unroll
  for (int nt = 0; nt < 8; ++nt) {
#pragma unroll
    for (int r = 0; r < 8; ++r) {
      int rr = rbase + r;
      if (rr < N)
        hs[(size_t)rr * (HEADS * HIDDEN) + st * 128 + nt * 16 + colb] = (half_t)acc[nt][r];
    }
  }
}

// ---------- attention scalars per node (one thread per node, all 4 heads) ----------
__global__ void k_att_scores(const half_t* __restrict__ h, const float* __restrict__ wsa,
                             const float* __restrict__ wda, float* __restrict__ a_s,
                             float* __restrict__ a_d, int N) {
  int n = blockIdx.x * blockDim.x + threadIdx.x;
  if (n >= N) return;
  const half_t* hr = h + (size_t)n * HIDDEN;
  float s0 = 0.f, s1 = 0.f, s2 = 0.f, s3 = 0.f;
  float d0 = 0.f, d1 = 0.f, d2 = 0.f, d3 = 0.f;
  for (int kk = 0; kk < HIDDEN; kk += 8) {
    v8h hv8 = *(const v8h*)(hr + kk);
#pragma unroll
    for (int j = 0; j < 8; ++j) {
      float hv = (float)hv8[j];
      float4 ws = *(const float4*)(wsa + (kk + j) * 4);
      float4 wd = *(const float4*)(wda + (kk + j) * 4);
      s0 += hv * ws.x; s1 += hv * ws.y; s2 += hv * ws.z; s3 += hv * ws.w;
      d0 += hv * wd.x; d1 += hv * wd.y; d2 += hv * wd.z; d3 += hv * wd.w;
    }
  }
  *(float4*)(a_s + (size_t)n * 4) = make_float4(s0, s1, s2, s3);
  *(float4*)(a_d + (size_t)n * 4) = make_float4(d0, d1, d2, d3);
}

// ---------- edge pass 1: logits + segment max ----------
__global__ void k_edge_logits(const int* __restrict__ src, const int* __restrict__ dst,
                              const float* __restrict__ a_s, const float* __restrict__ a_d,
                              float* __restrict__ e_edge, unsigned* __restrict__ emax, int E) {
  int e = blockIdx.x * blockDim.x + threadIdx.x;
  if (e >= E) return;
  int s = src[e], d = dst[e];
#pragma unroll
  for (int h = 0; h < HEADS; ++h) {
    float v = a_s[s * HEADS + h] + a_d[d * HEADS + h];
    v = (v > 0.f) ? v : 0.2f * v;   // leaky_relu(0.2)
    e_edge[(size_t)e * HEADS + h] = v;
    atomicMax(&emax[d * HEADS + h], enc_f32(v));
  }
}

// ---------- edge pass 2: exp + denom ----------
__global__ void k_edge_exp(const int* __restrict__ dst, float* __restrict__ e_edge,
                           const unsigned* __restrict__ emax, float* __restrict__ denom, int E) {
  int e = blockIdx.x * blockDim.x + threadIdx.x;
  if (e >= E) return;
  int d = dst[e];
#pragma unroll
  for (int h = 0; h < HEADS; ++h) {
    float m = dec_f32(emax[d * HEADS + h]);
    float ee = __expf(e_edge[(size_t)e * HEADS + h] - m);
    e_edge[(size_t)e * HEADS + h] = ee;
    atomicAdd(&denom[d * HEADS + h], ee);
  }
}

// ---------- edge pass 3: head-folded weighted scatter (wave per edge) ----------
__global__ void k_gat_scatter(const int* __restrict__ src, const int* __restrict__ dst,
                              const float* __restrict__ e_exp, const float* __restrict__ denom,
                              const half_t* __restrict__ hs, float* __restrict__ accum, int E) {
  int tid = blockIdx.x * blockDim.x + threadIdx.x;
  int e = tid >> 5, lane = tid & 31;
  if (e >= E) return;
  int s = src[e], d = dst[e];
  float alpha[HEADS];
#pragma unroll
  for (int h = 0; h < HEADS; ++h)
    alpha[h] = e_exp[(size_t)e * HEADS + h] / denom[d * HEADS + h];
  int c0 = lane * 4;
  float acc4[4] = {0.f, 0.f, 0.f, 0.f};
#pragma unroll
  for (int h = 0; h < HEADS; ++h) {
    v4h hv = *(const v4h*)(hs + ((size_t)s * HEADS + h) * HIDDEN + c0);
#pragma unroll
    for (int j = 0; j < 4; ++j) acc4[j] += alpha[h] * (float)hv[j];
  }
  float* ap = accum + (size_t)d * HIDDEN + c0;
#pragma unroll
  for (int j = 0; j < 4; ++j) atomicAdd(&ap[j], acc4[j]);
}

// ---------- final GEMM: y = relu(accum/4 + b_gat) @ W_lin + b_lin ----------
__global__ void __launch_bounds__(256) k_final_gemm(
    const float* __restrict__ accum, const float* __restrict__ b_gat,
    const half_t* __restrict__ wlin_p, const float* __restrict__ b_lin,
    float* __restrict__ y, int N) {
  __shared__ v8h sm[256];                // 4 tiles of current ks
  int tid = threadIdx.x;
  int waveid = tid >> 5, lane = tid & 31;
  int rt = blockIdx.x * 8 + waveid;
  int row0 = rt << 4;
  int g = lane >> 4;
  int rA = row0 + (lane & 15);
  if (rA >= N) rA = N - 1;
  const float* arow = accum + (size_t)rA * HIDDEN;

  v8f acc[4];
#pragma unroll
  for (int nt = 0; nt < 4; ++nt)
#pragma unroll
    for (int r = 0; r < 8; ++r) acc[nt][r] = 0.0f;

  for (int ks = 0; ks < 4; ++ks) {
    __syncthreads();
    const v8h* src = (const v8h*)wlin_p + ks * 256;
    sm[tid] = src[tid];
    __syncthreads();

    int kb = ks * 32 + g * 8;
    float fl[16];
    *(float4*)&fl[0]  = *(const float4*)(arow + kb);
    *(float4*)&fl[4]  = *(const float4*)(arow + kb + 4);
    *(float4*)&fl[8]  = *(const float4*)(arow + kb + 16);
    *(float4*)&fl[12] = *(const float4*)(arow + kb + 20);
    v16h a;
#pragma unroll
    for (int i = 0; i < 8; ++i) {
      float v0 = fl[i]     * 0.25f + b_gat[kb + i];
      float v1 = fl[8 + i] * 0.25f + b_gat[kb + 16 + i];
      a[i]     = (half_t)fmaxf(v0, 0.0f);
      a[8 + i] = (half_t)fmaxf(v1, 0.0f);
    }
#pragma unroll
    for (int nt = 0; nt < 4; ++nt) {
      int bi = nt * 64 + lane * 2;
      v16h b = SHUF16(sm[bi], sm[bi + 1]);
      acc[nt] = wmma_f16(a, b, acc[nt]);
    }
  }

  int colb = lane & 15;
  int rbase = row0 + g * 8;
#pragma unroll
  for (int nt = 0; nt < 4; ++nt) {
    int col = nt * 16 + colb;
    float bias = b_lin[col];
#pragma unroll
    for (int r = 0; r < 8; ++r) {
      int rr = rbase + r;
      if (rr < N) y[(size_t)rr * OUTD + col] = acc[nt][r] + bias;
    }
  }
}

extern "C" void kernel_launch(void* const* d_in, const int* in_sizes, int n_in,
                              void* d_out, int out_size, void* d_ws, size_t ws_size,
                              hipStream_t stream) {
  const float* x      = (const float*)d_in[0];
  const int*   ei     = (const int*)d_in[1];
  const float* W_l    = (const float*)d_in[2];
  const float* W_r    = (const float*)d_in[3];
  const float* b_sage = (const float*)d_in[4];
  const float* W_src  = (const float*)d_in[5];
  const float* W_dst  = (const float*)d_in[6];
  const float* att_s  = (const float*)d_in[7];
  const float* att_d  = (const float*)d_in[8];
  const float* b_gat  = (const float*)d_in[9];
  const float* W_lin  = (const float*)d_in[10];
  const float* b_lin  = (const float*)d_in[11];

  const int N = in_sizes[0] / D_FEAT;
  const int E = in_sizes[1] / 2;
  const int* src = ei;
  const int* dst = ei + E;

  // workspace carve-up (256B aligned regions)
  char* w = (char*)d_ws;
  auto alloc = [&](size_t bytes) -> char* {
    char* p = w;
    w += (bytes + 255) & ~(size_t)255;
    return p;
  };
  float*    agg    = (float*)alloc((size_t)N * HIDDEN * 4);   // reused as GAT accumulator
  float*    cnt    = (float*)alloc((size_t)N * 4);
  float*    a_s    = (float*)alloc((size_t)N * HEADS * 4);
  float*    a_d    = (float*)alloc((size_t)N * HEADS * 4);
  float*    denom  = (float*)alloc((size_t)N * HEADS * 4);
  unsigned* emax   = (unsigned*)alloc((size_t)N * HEADS * 4);
  float*    eearr  = (float*)alloc((size_t)E * HEADS * 4);
  half_t*   x_h    = (half_t*)alloc((size_t)N * D_FEAT * 2);
  half_t*   h_h    = (half_t*)alloc((size_t)N * HIDDEN * 2);
  half_t*   hs_h   = (half_t*)alloc((size_t)N * HEADS * HIDDEN * 2);
  half_t*   wl_p   = (half_t*)alloc((size_t)16384 * 2);       // packed B fragments
  half_t*   wr_p   = (half_t*)alloc((size_t)16384 * 2);
  half_t*   wsrc_p = (half_t*)alloc((size_t)65536 * 2);       // strip-major: [st][ks][nt][32][16]
  half_t*   wlin_p = (half_t*)alloc((size_t)8192 * 2);
  float*    wsa    = (float*)alloc((size_t)HIDDEN * HEADS * 4);
  float*    wda    = (float*)alloc((size_t)HIDDEN * HEADS * 4);

  const int B = 256;
  const int nrt = (N + 15) / 16;

  // init accumulators
  k_fill_f32<<<2048, B, 0, stream>>>(agg, 0.f, (long)N * HIDDEN);
  k_fill_f32<<<256, B, 0, stream>>>(cnt, 0.f, (long)N);
  k_fill_f32<<<256, B, 0, stream>>>(denom, 0.f, (long)N * HEADS);
  k_fill_u32<<<256, B, 0, stream>>>(emax, host_enc_f32(-1e30f), (long)N * HEADS);

  // weight prep: pack into WMMA B-fragment layout (fused f32->f16)
  k_pack_b<<<64, B, 0, stream>>>(W_l, HIDDEN, 0, HIDDEN, wl_p);
  k_pack_b<<<64, B, 0, stream>>>(W_r, HIDDEN, 0, HIDDEN, wr_p);
  for (int st = 0; st < 4; ++st)
    k_pack_b<<<64, B, 0, stream>>>(W_src, HEADS * HIDDEN, st * 128, 128, wsrc_p + (size_t)st * 16384);
  k_pack_b<<<32, B, 0, stream>>>(W_lin, OUTD, 0, OUTD, wlin_p);
  k_att_combine<<<1, 512, 0, stream>>>(W_src, W_dst, att_s, att_d, wsa, wda);
  k_cast_f16<<<2048, B, 0, stream>>>(x, x_h, (long)N * D_FEAT);

  // SAGE
  k_sage_scatter<<<(int)(((long)E * 32 + B - 1) / B), B, 0, stream>>>(x, src, dst, agg, cnt, E);
  k_sage_gemm<<<(nrt + 7) / 8, B, 0, stream>>>(agg, cnt, x_h, wl_p, wr_p, b_sage, h_h, N);

  // re-zero accumulator buffer for GAT (aliases agg)
  k_fill_f32<<<2048, B, 0, stream>>>(agg, 0.f, (long)N * HIDDEN);

  // GAT
  k_att_scores<<<(N + B - 1) / B, B, 0, stream>>>(h_h, wsa, wda, a_s, a_d, N);
  k_gat_gemm<<<((nrt + 7) / 8) * 4, B, 0, stream>>>(h_h, wsrc_p, hs_h, N);
  k_edge_logits<<<(E + B - 1) / B, B, 0, stream>>>(src, dst, a_s, a_d, eearr, emax, E);
  k_edge_exp<<<(E + B - 1) / B, B, 0, stream>>>(dst, eearr, emax, denom, E);
  k_gat_scatter<<<(int)(((long)E * 32 + B - 1) / B), B, 0, stream>>>(src, dst, eearr, denom, hs_h, agg, E);

  // final linear
  k_final_gemm<<<(nrt + 7) / 8, B, 0, stream>>>(agg, b_gat, wlin_p, b_lin, (float*)d_out, N);
}
